// OCRLayer_31834297598795
// MI455X (gfx1250) — matrix operand order repaired and checked by reading, not compiled, not run on hardware
//
#include <hip/hip_runtime.h>

// ---------------- constants (structural to the reference) ----------------
#define C_DIM   96
#define KC_DIM  128
#define K_CLS   20
#define INC_DIM 64
#define BATCH   8
#define INVS    0.99999500003749977f   /* 1/sqrt(1+1e-5) */
#define INVSQ   0.08838834764831845f   /* 128^-0.5 */

typedef __attribute__((ext_vector_type(16))) _Float16 v16h;
typedef __attribute__((ext_vector_type(8)))  float    v8f;

// wave-local LDS fence: DS ops are in-order per wave; this stops the
// compiler/scheduler from moving VALU uses above outstanding DS ops.
__device__ __forceinline__ void wfence() {
  __builtin_amdgcn_wave_barrier();
  asm volatile("s_wait_dscnt 0x0" ::: "memory");
  __builtin_amdgcn_wave_barrier();
}

// A-fragment (16x32 f16, MxK) from a row-major f32 LDS tile.
// Lane L: m=L&15, h=L>>4. half j: k = k0 + h*16 + j.
__device__ __forceinline__ v16h ldsA(const float* buf, int stride, int lane,
                                     int k0, int kmax) {
  int m = lane & 15, h = lane >> 4;
  v16h a;
#pragma unroll
  for (int j = 0; j < 16; ++j) {
    int k = k0 + h * 16 + j;
    a[j] = (_Float16)((k < kmax) ? buf[m * stride + k] : 0.f);
  }
  return a;
}

// Pre-packed B fragment: 512 halves, lane L owns halves [L*16, L*16+16).
__device__ __forceinline__ v16h ldB(const _Float16* frag, int lane) {
  return *(const v16h*)(frag + lane * 16);
}

__device__ __forceinline__ v8f wmma(v16h a, v16h b, v8f c) {
  return __builtin_amdgcn_wmma_f32_16x16x32_f16(false, a, false, b,
                                                (short)0, c, false, false);
}

// ---------------- phase 0: zero accumulators ----------------
__global__ void k_init(float* num, float* sum, int* starts, int* ends, int N) {
  for (int i = threadIdx.x; i < BATCH * K_CLS * C_DIM; i += blockDim.x) num[i] = 0.f;
  for (int i = threadIdx.x; i < BATCH * K_CLS; i += blockDim.x) sum[i] = 0.f;
  if (threadIdx.x < BATCH) { starts[threadIdx.x] = N; ends[threadIdx.x] = 0; }
}

// ---------------- phase 1: batch ranges (indices sorted) ----------------
__global__ void k_bounds(const int* __restrict__ bi, int* starts, int* ends, int N) {
  int n = blockIdx.x * blockDim.x + threadIdx.x;
  if (n >= N) return;
  int b = bi[n];
  if (b < 0 || b >= BATCH) return;
  if (n == 0 || bi[n - 1] != b) atomicMin(&starts[b], n);
  if (n == N - 1 || bi[n + 1] != b) atomicMax(&ends[b], n + 1);
}

// ---------------- phase 2: gather numerator + sumexp -------------------
// num[b][k][c] = sum_n exp(probs[n][k]) * feats[n][c];  sumexp[b][k] = sum exp
__global__ void k_accum(const float* __restrict__ feats,
                        const float* __restrict__ probs,
                        const int* __restrict__ starts, const int* __restrict__ ends,
                        float* num, float* sumexp, int N) {
  int b = blockIdx.y;
  int s = starts[b], e = ends[b];
  if (s >= e) return;
  int nch = gridDim.x;
  int len = (e - s + nch - 1) / nch;
  int cs = s + blockIdx.x * len;
  int ce = min(cs + len, e);
  if (cs >= ce) return;

  __shared__ float pe[64 * K_CLS];
  __shared__ float fl[64 * C_DIM];
  int t = threadIdx.x;
  int k = t / 12, cseg = t % 12;   // 240 worker threads: 20 classes x 12 col-chunks
  float acc[8] = {0, 0, 0, 0, 0, 0, 0, 0};
  float accs = 0.f;

  for (int p0 = cs; p0 < ce; p0 += 64) {
    int cnt = min(64, ce - p0);
    for (int i = t; i < 64 * K_CLS; i += 256) {
      int p = i / K_CLS, kk = i % K_CLS;
      pe[i] = (p < cnt) ? __expf(probs[(p0 + p) * K_CLS + kk]) : 0.f;
    }
    for (int i = t; i < 64 * 24; i += 256) {   // 96 floats = 24 float4 per row
      int p = i / 24, c4 = i % 24;
      float4 v4 = (p < cnt) ? ((const float4*)(feats + (size_t)(p0 + p) * C_DIM))[c4]
                            : make_float4(0.f, 0.f, 0.f, 0.f);
      ((float4*)fl)[p * 24 + c4] = v4;
    }
    __syncthreads();
    if (t < 240) {
      for (int p = 0; p < 64; ++p) {
        float ev = pe[p * K_CLS + k];
        accs += ev;
#pragma unroll
        for (int j = 0; j < 8; ++j)
          acc[j] += ev * fl[p * C_DIM + cseg * 8 + j];
      }
    }
    __syncthreads();
  }
  if (t < 240) {
#pragma unroll
    for (int j = 0; j < 8; ++j)
      atomicAdd(&num[(b * K_CLS + k) * C_DIM + cseg * 8 + j], acc[j]);
    if (cseg == 0) atomicAdd(&sumexp[b * K_CLS + k], accs);
  }
}

// ---------------- phase 3: keys/vals + fragment packing ----------------
// blocks 0..7: per-batch keys/vals (f32 copy + B-fragment pack)
// block 8: pack Wq (3x8 frags) and Wb (4x6);  block 9: pack Wo (6x4)
__global__ void k_prep(const float* __restrict__ num, const float* __restrict__ sumexp,
                       const float* __restrict__ Wk, const float* __restrict__ gk,
                       const float* __restrict__ bk,
                       const float* __restrict__ Wv, const float* __restrict__ gv,
                       const float* __restrict__ bv,
                       const float* __restrict__ Wq, const float* __restrict__ Wb,
                       const float* __restrict__ Wo,
                       float* keysF, float* valsF,
                       _Float16* keysB, _Float16* valsB,
                       _Float16* WqB, _Float16* WbB, _Float16* WoB) {
  int blk = blockIdx.x;
  if (blk < BATCH) {
    __shared__ float kv[2 * K_CLS * KC_DIM];   // keys | vals
    int b = blk;
    for (int idx = threadIdx.x; idx < K_CLS * KC_DIM; idx += blockDim.x) {
      int k = idx >> 7, kc = idx & 127;
      float se = sumexp[b * K_CLS + k];
      float inv = (se > 0.f) ? 1.f / se : 0.f;
      float aK = 0.f, aV = 0.f;
      for (int c = 0; c < C_DIM; ++c) {
        float cv = num[(b * K_CLS + k) * C_DIM + c] * inv;
        aK += cv * Wk[c * KC_DIM + kc];
        aV += cv * Wv[c * KC_DIM + kc];
      }
      float kval = fmaxf(0.f, aK * gk[kc] * INVS + bk[kc]);
      float vval = fmaxf(0.f, aV * gv[kc] * INVS + bv[kc]);
      kv[idx] = kval; kv[K_CLS * KC_DIM + idx] = vval;
      keysF[b * K_CLS * KC_DIM + idx] = kval;
      valsF[b * K_CLS * KC_DIM + idx] = vval;
    }
    __syncthreads();
    // keysB: B = keys^T (K=kc 128 -> 4 ktiles, N=class pad32 -> 2 ntiles)
    // valsB: B = vals   (K=class pad32 -> 1 ktile, N=kc -> 8 ntiles)
    for (int idx = threadIdx.x; idx < 8 * 512; idx += blockDim.x) {
      int frag = idx >> 9, e2 = idx & 511, ln = e2 >> 4, j = e2 & 15;
      int kt = frag >> 1, nt = frag & 1;
      int kk = kt * 32 + (ln >> 4) * 16 + j;          // kc
      int nn = nt * 16 + (ln & 15);                   // class
      keysB[b * 4096 + idx] =
          (_Float16)((nn < K_CLS) ? kv[nn * KC_DIM + kk] : 0.f);
      int kcls = (ln >> 4) * 16 + j;                  // class (pad 32)
      int nkc = frag * 16 + (ln & 15);                // kc
      valsB[b * 4096 + idx] =
          (_Float16)((kcls < K_CLS) ? kv[K_CLS * KC_DIM + kcls * KC_DIM + nkc] : 0.f);
    }
  } else if (blk == BATCH) {
    for (int idx = threadIdx.x; idx < 24 * 512; idx += blockDim.x) {
      int frag = idx >> 9, e2 = idx & 511, ln = e2 >> 4, j = e2 & 15;
      { // Wq [96][128] : 3 ktiles x 8 ntiles
        int kt = frag >> 3, nt = frag & 7;
        int k = kt * 32 + (ln >> 4) * 16 + j, n = nt * 16 + (ln & 15);
        WqB[idx] = (_Float16)Wq[k * KC_DIM + n];
      }
      { // Wb [128][96] : 4 ktiles x 6 ntiles
        int kt = frag / 6, nt = frag % 6;
        int k = kt * 32 + (ln >> 4) * 16 + j, n = nt * 16 + (ln & 15);
        WbB[idx] = (_Float16)Wb[k * C_DIM + n];
      }
    }
  } else {
    for (int idx = threadIdx.x; idx < 24 * 512; idx += blockDim.x) {
      int frag = idx >> 9, e2 = idx & 511, ln = e2 >> 4, j = e2 & 15;
      int kt = frag >> 2, nt = frag & 3;              // Wo [192][64]: 6 x 4
      int k = kt * 32 + (ln >> 4) * 16 + j, n = nt * 16 + (ln & 15);
      WoB[idx] = (_Float16)Wo[k * INC_DIM + n];
    }
  }
}

// ---------------- phase 4: fused per-tile WMMA pipeline ----------------
__global__ __launch_bounds__(128) void k_main(
    const float* __restrict__ feats, const int* __restrict__ bi,
    const float* __restrict__ gq, const float* __restrict__ bq,
    const float* __restrict__ gbv, const float* __restrict__ bbv,
    const float* __restrict__ gov, const float* __restrict__ bov,
    const _Float16* __restrict__ keysB, const _Float16* __restrict__ valsB,
    const _Float16* __restrict__ WqB, const _Float16* __restrict__ WbB,
    const _Float16* __restrict__ WoB,
    const float* __restrict__ keysF, const float* __restrict__ valsF,
    float* __restrict__ out, int N, int numTiles) {
  __shared__ float smem[4 * 2688];          // per wave: 16x132 tile + 16x36 sim
  int lane = threadIdx.x & 31;
  int wv = threadIdx.x >> 5;
  int tile = blockIdx.x * 4 + wv;
  if (tile >= numTiles) return;             // wave-uniform: EXEC stays full
  float* buf = smem + wv * 2688;            // 16 x 132 f32
  float* simb = buf + 2112;                 // 16 x 36 f32
  int row0 = tile * 16;
  int h = lane >> 4, nl = lane & 15;

  // stage feats tile (16 x 96) into LDS, coalesced float4
  for (int i = lane; i < 16 * 24; i += 32) {
    int r = i / 24, c4 = (i % 24) * 4;
    int row = min(row0 + r, N - 1);
    float4 f4 = *(const float4*)(feats + (size_t)row * C_DIM + c4);
    *(float4*)(buf + r * 132 + c4) = f4;
  }
  wfence();
  v16h fa0 = ldsA(buf, 132, lane, 0, 96);
  v16h fa1 = ldsA(buf, 132, lane, 32, 96);
  v16h fa2 = ldsA(buf, 132, lane, 64, 96);
  wfence();

  // Q = relu(bn(feats @ Wq))  -> buf[16][128]
#pragma unroll
  for (int nt = 0; nt < 8; ++nt) {
    v8f c = {};
    c = wmma(fa0, ldB(WqB + (0 * 8 + nt) * 512, lane), c);
    c = wmma(fa1, ldB(WqB + (1 * 8 + nt) * 512, lane), c);
    c = wmma(fa2, ldB(WqB + (2 * 8 + nt) * 512, lane), c);
    int n = nt * 16 + nl;
    float s = gq[n] * INVS, bia = bq[n];
#pragma unroll
    for (int r = 0; r < 8; ++r)
      buf[(r + 8 * h) * 132 + n] = fmaxf(0.f, c[r] * s + bia);
  }
  wfence();
  v16h q0 = ldsA(buf, 132, lane, 0, 128);
  v16h q1 = ldsA(buf, 132, lane, 32, 128);
  v16h q2 = ldsA(buf, 132, lane, 64, 128);
  v16h q3 = ldsA(buf, 132, lane, 96, 128);

  int b0 = bi[row0];
  int b1 = bi[min(row0 + 15, N - 1)];
  if (b0 == b1) {
    // sim = Q @ keys^T * 128^-0.5 -> simb[16][32]
#pragma unroll
    for (int nt = 0; nt < 2; ++nt) {
      v8f c = {};
      c = wmma(q0, ldB(keysB + (b0 * 8 + 0 * 2 + nt) * 512, lane), c);
      c = wmma(q1, ldB(keysB + (b0 * 8 + 1 * 2 + nt) * 512, lane), c);
      c = wmma(q2, ldB(keysB + (b0 * 8 + 2 * 2 + nt) * 512, lane), c);
      c = wmma(q3, ldB(keysB + (b0 * 8 + 3 * 2 + nt) * 512, lane), c);
#pragma unroll
      for (int r = 0; r < 8; ++r)
        simb[(r + 8 * h) * 36 + nt * 16 + nl] = c[r] * INVSQ;
    }
    wfence();
    if (lane < 16) {                        // row-wise softmax over 20 classes
      float mx = -1e30f;
#pragma unroll
      for (int k = 0; k < K_CLS; ++k) mx = fmaxf(mx, simb[lane * 36 + k]);
      float ss = 0.f;
#pragma unroll
      for (int k = 0; k < K_CLS; ++k) {
        float e = __expf(simb[lane * 36 + k] - mx);
        ss += e; simb[lane * 36 + k] = e;
      }
      float inv = 1.f / ss;
#pragma unroll
      for (int k = 0; k < K_CLS; ++k) simb[lane * 36 + k] *= inv;
    }
    wfence();
    v16h av = ldsA(simb, 36, lane, 0, K_CLS);   // kmax=20 masks pad classes
#pragma unroll
    for (int nt = 0; nt < 8; ++nt) {
      v8f c = {};
      c = wmma(av, ldB(valsB + (b0 * 8 + nt) * 512, lane), c);
#pragma unroll
      for (int r = 0; r < 8; ++r)
        buf[(r + 8 * h) * 132 + nt * 16 + nl] = c[r];
    }
  } else {
    // rare batch-boundary tile: per-row scalar attention
    if (lane < 16) {
      int row = min(row0 + lane, N - 1);
      int bb2 = bi[row];
      const float* qrow = buf + lane * 132;
      float sim[K_CLS];
      float mx = -1e30f;
#pragma unroll 4
      for (int k = 0; k < K_CLS; ++k) {
        const float* kr = keysF + (bb2 * K_CLS + k) * KC_DIM;
        float acc = 0.f;
        for (int kc = 0; kc < KC_DIM; ++kc) acc += qrow[kc] * kr[kc];
        acc *= INVSQ;
        sim[k] = acc; mx = fmaxf(mx, acc);
      }
      float ss = 0.f;
#pragma unroll
      for (int k = 0; k < K_CLS; ++k) { sim[k] = __expf(sim[k] - mx); ss += sim[k]; }
      float inv = 1.f / ss;
#pragma unroll
      for (int k = 0; k < K_CLS; ++k) sim[k] *= inv;
      for (int kc = 0; kc < KC_DIM; ++kc) {
        float acc = 0.f;
#pragma unroll
        for (int k = 0; k < K_CLS; ++k)
          acc += sim[k] * valsF[(bb2 * K_CLS + k) * KC_DIM + kc];
        buf[lane * 132 + kc] = acc;
      }
    }
  }
  wfence();
  v16h o0 = ldsA(buf, 132, lane, 0, 128);
  v16h o1 = ldsA(buf, 132, lane, 32, 128);
  v16h o2 = ldsA(buf, 132, lane, 64, 128);
  v16h o3 = ldsA(buf, 132, lane, 96, 128);
  wfence();

  // context = relu(bn(out @ Wb)) -> buf[16][96]
#pragma unroll
  for (int nt = 0; nt < 6; ++nt) {
    v8f c = {};
    c = wmma(o0, ldB(WbB + (0 * 6 + nt) * 512, lane), c);
    c = wmma(o1, ldB(WbB + (1 * 6 + nt) * 512, lane), c);
    c = wmma(o2, ldB(WbB + (2 * 6 + nt) * 512, lane), c);
    c = wmma(o3, ldB(WbB + (3 * 6 + nt) * 512, lane), c);
    int n = nt * 16 + nl;
    float s = gbv[n] * INVS, bia = bbv[n];
#pragma unroll
    for (int r = 0; r < 8; ++r)
      buf[(r + 8 * h) * 132 + n] = fmaxf(0.f, c[r] * s + bia);
  }
  wfence();
  v16h c0 = ldsA(buf, 132, lane, 0, 96);
  v16h c1 = ldsA(buf, 132, lane, 32, 96);
  v16h c2 = ldsA(buf, 132, lane, 64, 96);

  // final: relu(bn([context, feats] @ Wo)) -> out[N][64]
#pragma unroll
  for (int nt = 0; nt < 4; ++nt) {
    v8f c = {};
    c = wmma(c0,  ldB(WoB + (0 * 4 + nt) * 512, lane), c);
    c = wmma(c1,  ldB(WoB + (1 * 4 + nt) * 512, lane), c);
    c = wmma(c2,  ldB(WoB + (2 * 4 + nt) * 512, lane), c);
    c = wmma(fa0, ldB(WoB + (3 * 4 + nt) * 512, lane), c);
    c = wmma(fa1, ldB(WoB + (4 * 4 + nt) * 512, lane), c);
    c = wmma(fa2, ldB(WoB + (5 * 4 + nt) * 512, lane), c);
    int n = nt * 16 + nl;
    float s = gov[n] * INVS, bia = bov[n];
#pragma unroll
    for (int r = 0; r < 8; ++r) {
      int row = row0 + r + 8 * h;
      if (row < N) out[(size_t)row * INC_DIM + n] = fmaxf(0.f, c[r] * s + bia);
    }
  }
}

// ---------------- host launcher ----------------
extern "C" void kernel_launch(void* const* d_in, const int* in_sizes, int n_in,
                              void* d_out, int out_size, void* d_ws, size_t ws_size,
                              hipStream_t stream) {
  const float* feats = (const float*)d_in[0];
  const float* probs = (const float*)d_in[1];
  const int*   bi    = (const int*)d_in[2];
  const float* Wq = (const float*)d_in[4];
  const float* gq = (const float*)d_in[5];
  const float* bq = (const float*)d_in[6];
  const float* Wk = (const float*)d_in[7];
  const float* gk = (const float*)d_in[8];
  const float* bk = (const float*)d_in[9];
  const float* Wv = (const float*)d_in[10];
  const float* gv = (const float*)d_in[11];
  const float* bv = (const float*)d_in[12];
  const float* Wb = (const float*)d_in[13];
  const float* gb = (const float*)d_in[14];
  const float* bb = (const float*)d_in[15];
  const float* Wo = (const float*)d_in[16];
  const float* go = (const float*)d_in[17];
  const float* bo = (const float*)d_in[18];
  float* outp = (float*)d_out;
  int N = in_sizes[0] / C_DIM;

  // workspace layout (floats, then halves, then ints)
  float* wsf      = (float*)d_ws;
  float* d_num    = wsf;                 // 8*20*96 = 15360
  float* d_sum    = wsf + 15360;         // 160
  float* d_keysF  = wsf + 15520;         // 20480
  float* d_valsF  = wsf + 36000;         // 20480
  _Float16* h0    = (_Float16*)(wsf + 56480);
  _Float16* d_keysB = h0;                // 8*8*512 = 32768
  _Float16* d_valsB = h0 + 32768;        // 32768
  _Float16* d_WqB   = h0 + 65536;        // 12288
  _Float16* d_WbB   = h0 + 77824;        // 12288
  _Float16* d_WoB   = h0 + 90112;        // 12288
  int* d_starts     = (int*)(h0 + 102400);
  int* d_ends       = d_starts + BATCH;

  k_init<<<1, 256, 0, stream>>>(d_num, d_sum, d_starts, d_ends, N);
  k_bounds<<<(N + 255) / 256, 256, 0, stream>>>(bi, d_starts, d_ends, N);
  k_accum<<<dim3(160, BATCH), 256, 0, stream>>>(feats, probs, d_starts, d_ends,
                                                d_num, d_sum, N);
  k_prep<<<BATCH + 2, 256, 0, stream>>>(d_num, d_sum, Wk, gk, bk, Wv, gv, bv,
                                        Wq, Wb, Wo, d_keysF, d_valsF,
                                        d_keysB, d_valsB, d_WqB, d_WbB, d_WoB);
  int numTiles = (N + 15) / 16;
  k_main<<<(numTiles + 3) / 4, 128, 0, stream>>>(
      feats, bi, gq, bq, gb, bb, go, bo,
      d_keysB, d_valsB, d_WqB, d_WbB, d_WoB,
      d_keysF, d_valsF, outp, N, numTiles);
}
